// SuperModel_40054865003181
// MI455X (gfx1250) — compile-verified, matrix operand order checked by reading, not run on hardware
//
#include <hip/hip_runtime.h>
#include <stdint.h>

// ---------------------------------------------------------------------------
// CDNA5 / gfx1250 implementation.
//   Stage 1: fused elementwise front-end -> x[512 x 20064] bf16 (zero padded)
//   Stage 2-5: bf16 WMMA GEMMs (v_wmma_f32_16x16x32_bf16), ping-pong LDS
//              pipeline, async global->LDS copy for the bf16 A tiles,
//              fp32 weights converted to bf16 (v_cvt_pk_bf16_f32) while
//              staging into LDS.
//   Stage 6: tiny 78->2 fp32 output layer.
//
// Guard philosophy: activation K-padding is written as zeros and the GEMM
// epilogue zeroes N-padding columns, so out-of-range weights only ever
// multiply zeros. Loads therefore need only index CLAMPING (memory safety),
// never value masking -> completely branchless staging.
// ---------------------------------------------------------------------------

typedef __attribute__((ext_vector_type(16))) __bf16 v16bf;
typedef __attribute__((ext_vector_type(8)))  float  v8f;
typedef int v4i_ __attribute__((ext_vector_type(4)));

#if __has_builtin(__builtin_amdgcn_global_load_async_to_lds_b128)
#define HAVE_ASYNC_LDS 1
typedef __attribute__((address_space(1))) v4i_* as1_v4i;
typedef __attribute__((address_space(3))) v4i_* as3_v4i;
__device__ __forceinline__ void async_copy16B(const unsigned short* g, unsigned short* l) {
  __builtin_amdgcn_global_load_async_to_lds_b128((as1_v4i)g, (as3_v4i)l, 0, 0);
}
#else
#define HAVE_ASYNC_LDS 0
#endif

__device__ __forceinline__ void wait_async0() {
#if HAVE_ASYNC_LDS
#if __has_builtin(__builtin_amdgcn_s_wait_asynccnt)
  __builtin_amdgcn_s_wait_asynccnt(0);
#else
  asm volatile("s_wait_asynccnt 0x0" ::: "memory");
#endif
#endif
}

__device__ __forceinline__ unsigned short f2bf(float f) {
  union { float f; uint32_t u; } c; c.f = f;
  uint32_t u = c.u;
  uint32_t r = (u + 0x7FFFu + ((u >> 16) & 1u)) >> 16;  // round-to-nearest-even
  return (unsigned short)r;
}
__device__ __forceinline__ float bf2f(unsigned short s) {
  union { uint32_t u; float f; } c; c.u = ((uint32_t)s) << 16;
  return c.f;
}
// Packed fp32->bf16 RNE: single v_cvt_pk_bf16_f32 when available.
__device__ __forceinline__ unsigned int pack2bf(float a, float b) {
#if __has_builtin(__builtin_amdgcn_cvt_pk_bf16_f32)
  typedef __attribute__((ext_vector_type(2))) __bf16 v2bf;
  union { v2bf v; unsigned int u; } c;
  c.v = __builtin_amdgcn_cvt_pk_bf16_f32(a, b);
  return c.u;
#else
  return (unsigned int)f2bf(a) | ((unsigned int)f2bf(b) << 16);
#endif
}

// ---------------------------------------------------------------------------
// Kernel 1: fused gene-MLP + T-reduce + clinical embeddings -> bf16 x buffer
// x layout: [512][20064], cols 0..19999 = expr, 20000..20037 = clin, rest 0.
// ---------------------------------------------------------------------------
__global__ __launch_bounds__(256)
void build_x_kernel(const float* __restrict__ expr,
                    const int*   __restrict__ cc,
                    const float* __restrict__ cnc,
                    const float4* __restrict__ bw1,
                    const float4* __restrict__ bb1,
                    const float4* __restrict__ bw2,
                    const float* __restrict__ bb2,
                    const float* __restrict__ gw,
                    const float* __restrict__ gb,
                    const float* __restrict__ race,
                    const float* __restrict__ eth,
                    const float* __restrict__ inter,
                    const float* __restrict__ prot,
                    const float* __restrict__ mw,
                    const float* __restrict__ mb,
                    unsigned short* __restrict__ X)
{
  const int g = blockIdx.x * 256 + threadIdx.x;
  const int b = blockIdx.y;
  if (g >= 20064) return;

  float val = 0.f;
  if (g < 20000) {
    float acc = 0.f;
#pragma unroll
    for (int t = 0; t < 2; ++t) {
      const int u = t * 20000 + g;
      const float xv = expr[b * 40000 + u];
      const float4 w1 = bw1[u];
      const float4 c1 = bb1[u];
      const float4 w2 = bw2[u];
      const float h0 = fmaxf(xv * w1.x + c1.x, 0.f);
      const float h1 = fmaxf(xv * w1.y + c1.y, 0.f);
      const float h2 = fmaxf(xv * w1.z + c1.z, 0.f);
      const float h3 = fmaxf(xv * w1.w + c1.w, 0.f);
      const float s  = fmaxf(h0 * w2.x + h1 * w2.y + h2 * w2.z + h3 * w2.w + bb2[u], 0.f);
      acc += s * gw[g * 2 + t];
    }
    val = fmaxf(acc + gb[g], 0.f);
  } else if (g < 20038) {
    const int j  = g - 20000;
    const int c0 = cc[b * 3 + 0];
    const int c1i = cc[b * 3 + 1];
    const int c2 = cc[b * 3 + 2];
    if (j < 8)        val = race[c0 * 8 + j];
    else if (j < 12)  val = eth[c1i * 4 + (j - 8)];
    else if (j < 20)  val = inter[(c0 * 4 + c1i) * 8 + (j - 12)];
    else if (j < 28)  val = prot[c2 * 8 + (j - 20)];
    else {
      const int v = j - 28;
      val = fmaxf(cnc[b * 20 + v] * mw[v * 2 + 0] +
                  cnc[b * 20 + 10 + v] * mw[v * 2 + 1] + mb[v], 0.f);
    }
  }
  X[(size_t)b * 20064 + g] = f2bf(val);
}

// ---------------------------------------------------------------------------
// Kernel 2: bf16 WMMA GEMM:  Out = relu(A[512 x K] @ W[K x N] + bias) (bf16)
//   Compile-time K/N so the 16 column loads fold into immediate offsets.
//   Block 256 thr = 8 waves; block tile 128x128; wave tile 32x64; BK=32.
//   Ping-pong LDS buffers, one barrier per K-step; A tile copied with
//   GLOBAL_LOAD_ASYNC_TO_LDS (bf16, no conversion needed), B tile loaded to
//   registers early and packed-converted fp32->bf16 while storing [n][k].
//   LDS rows padded to 80B so 16-lane b128 fragment reads are conflict-free.
// ---------------------------------------------------------------------------
#define LDSW 40   // ushorts per LDS row (32 data + 8 pad = 80 bytes)

template <int K, int KPAD, int N, int NPAD>
__global__ __launch_bounds__(256)
void gemm_bf16_relu(const unsigned short* __restrict__ A,
                    const float* __restrict__ W,
                    const float* __restrict__ bias,
                    unsigned short* __restrict__ Out)
{
  constexpr int KSTEPS = KPAD / 32;
  __shared__ unsigned short sA[2][128 * LDSW];
  __shared__ unsigned short sB[2][128 * LDSW];

  const int tid  = threadIdx.x;
  const int lane = tid & 31;
  const int wave = tid >> 5;
  const int wm = wave & 3;          // 4 waves along M  -> 32 rows each
  const int wn = wave >> 2;         // 2 waves along N  -> 64 cols each
  const int hi = lane >> 4;
  const int ln = lane & 15;
  const int m0 = blockIdx.x * 128;  // M-fastest grid: panels share weights in L2
  const int n0 = blockIdx.y * 128;

  // A staging: each thread moves 32B (16 bf16): row tid>>1, K-half tid&1
  const int a_m = tid >> 1;
  const int a_half = tid & 1;
  const unsigned short* aRow = A + (size_t)(m0 + a_m) * KPAD + a_half * 16;
  const int aLds = a_m * LDSW + a_half * 16;
  // B staging: each thread loads a 16-deep column of fp32 (coalesced across n)
  const int b_n = tid & 127;
  const int b_k = (tid >> 7) * 16;
  const int gn  = n0 + b_n;
  const int gnc = (gn < N) ? gn : (N - 1);   // clamp only; epilogue zeroes pad
  const int bLds = b_n * LDSW + b_k;

  uint4 aReg0, aReg1;   // fallback (non-async) A staging
  float bReg[16];

  // ---- per-tile staging helpers (branchless; only uniform scalar branch) --
  auto issueA = [&](int buf, int k0) {
#if HAVE_ASYNC_LDS
    const unsigned short* ga = aRow + k0;
    unsigned short* la = &sA[buf][aLds];
    async_copy16B(ga, la);
    async_copy16B(ga + 8, la + 8);
#else
    const uint4* src = (const uint4*)(aRow + k0);
    aReg0 = src[0];
    aReg1 = src[1];
    (void)buf;
#endif
  };
  auto commitA = [&](int buf) {
#if !HAVE_ASYNC_LDS
    uint4* dst = (uint4*)&sA[buf][aLds];
    dst[0] = aReg0;
    dst[1] = aReg1;
#else
    (void)buf;
#endif
  };
  auto issueB = [&](int k0) {
    if (k0 + 32 <= K) {            // uniform fast path: immediate-offset loads
      const float* wp = W + (size_t)(k0 + b_k) * N + gnc;
#pragma unroll
      for (int i = 0; i < 16; ++i) bReg[i] = wp[(size_t)i * N];
    } else {                       // single tail step: clamp k, no masking
#pragma unroll
      for (int i = 0; i < 16; ++i) {
        const int gk = k0 + b_k + i;
        const int gkc = gk < K ? gk : (K - 1);
        bReg[i] = W[(size_t)gkc * N + gnc];
      }
    }
  };
  auto commitB = [&](int buf) {
    unsigned int pk[8];
#pragma unroll
    for (int i = 0; i < 8; ++i)
      pk[i] = pack2bf(bReg[2 * i], bReg[2 * i + 1]);   // v_cvt_pk_bf16_f32
    uint4* dst = (uint4*)&sB[buf][bLds];
    dst[0] = make_uint4(pk[0], pk[1], pk[2], pk[3]);
    dst[1] = make_uint4(pk[4], pk[5], pk[6], pk[7]);
  };

  v8f vzero = {};
  v8f acc[2][4];
#pragma unroll
  for (int i = 0; i < 2; ++i)
#pragma unroll
    for (int j = 0; j < 4; ++j)
      acc[i][j] = vzero;

  // ---- prologue: tile 0 into buffer 0 ----
  issueA(0, 0);
  issueB(0);
  commitA(0);
  commitB(0);
  wait_async0();
  __syncthreads();

  // ---- main pipelined loop: one barrier per K-step ----
  for (int kt = 0; kt < KSTEPS; ++kt) {
    const int cur = kt & 1;
    const int nxt = cur ^ 1;
    const bool more = (kt + 1 < KSTEPS);
    if (more) {                    // overlap next tile's fetch with WMMAs
      issueA(nxt, (kt + 1) * 32);
      issueB((kt + 1) * 32);
    }
    // fragments per ISA 7.12.2: K chunks at elements hi*8 and 16+hi*8
    union Frag { uint4 q[2]; v16bf v; };
    Frag af[2], bfr[4];
#pragma unroll
    for (int i = 0; i < 2; ++i) {
      const unsigned short* base = &sA[cur][(wm * 32 + i * 16 + ln) * LDSW];
      af[i].q[0] = *(const uint4*)(base + hi * 8);
      af[i].q[1] = *(const uint4*)(base + 16 + hi * 8);
    }
#pragma unroll
    for (int j = 0; j < 4; ++j) {
      const unsigned short* base = &sB[cur][(wn * 64 + j * 16 + ln) * LDSW];
      bfr[j].q[0] = *(const uint4*)(base + hi * 8);
      bfr[j].q[1] = *(const uint4*)(base + 16 + hi * 8);
    }
#pragma unroll
    for (int i = 0; i < 2; ++i)
#pragma unroll
      for (int j = 0; j < 4; ++j)
        acc[i][j] = __builtin_amdgcn_wmma_f32_16x16x32_bf16(
            false, af[i].v, false, bfr[j].v, (short)0, acc[i][j], false, false);
    if (more) {
      commitA(nxt);
      commitB(nxt);
      wait_async0();
    }
    __syncthreads();
  }

  // ---- epilogue: bias + ReLU + packed bf16 store (zero into N padding) ----
#pragma unroll
  for (int j = 0; j < 4; ++j) {
    const int col = n0 + wn * 64 + j * 16 + ln;
    if (col >= NPAD) continue;
    const float bj = (col < N) ? bias[col] : 0.f;
#pragma unroll
    for (int i = 0; i < 2; ++i) {
#pragma unroll
      for (int r = 0; r < 8; r += 2) {   // convert row-pairs with one cvt_pk
        const int row = m0 + wm * 32 + i * 16 + r + 8 * hi;
        const float v0 = (col < N) ? fmaxf(acc[i][j][r] + bj, 0.f) : 0.f;
        const float v1 = (col < N) ? fmaxf(acc[i][j][r + 1] + bj, 0.f) : 0.f;
        const unsigned int p = pack2bf(v0, v1);
        Out[(size_t)row * NPAD + col]       = (unsigned short)p;
        Out[(size_t)(row + 1) * NPAD + col] = (unsigned short)(p >> 16);
      }
    }
  }
}

// ---------------------------------------------------------------------------
// Kernel 3: final 78 -> 2 layer, fp32 output (no relu).
// ---------------------------------------------------------------------------
__global__ __launch_bounds__(256)
void final_layer_kernel(const unsigned short* __restrict__ H,
                        const float* __restrict__ W,
                        const float* __restrict__ bias,
                        float* __restrict__ out)
{
  const int t = blockIdx.x * 256 + threadIdx.x;
  if (t >= 1024) return;
  const int b = t >> 1, o = t & 1;
  float s = bias[o];
#pragma unroll 13
  for (int k = 0; k < 78; ++k)
    s += bf2f(H[b * 96 + k]) * W[k * 2 + o];
  out[b * 2 + o] = s;
}

// ---------------------------------------------------------------------------
extern "C" void kernel_launch(void* const* d_in, const int* in_sizes, int n_in,
                              void* d_out, int out_size, void* d_ws, size_t ws_size,
                              hipStream_t stream) {
  (void)in_sizes; (void)n_in; (void)out_size; (void)ws_size;

  const float* expr = (const float*)d_in[0];
  const int*   cc   = (const int*)d_in[1];
  const float* cnc  = (const float*)d_in[2];
  const float4* bw1 = (const float4*)d_in[3];
  const float4* bb1 = (const float4*)d_in[4];
  const float4* bw2 = (const float4*)d_in[5];
  const float* bb2  = (const float*)d_in[6];
  const float* gw   = (const float*)d_in[7];
  const float* gb   = (const float*)d_in[8];
  const float* race = (const float*)d_in[9];
  const float* eth  = (const float*)d_in[10];
  const float* inter= (const float*)d_in[11];
  const float* prot = (const float*)d_in[12];
  const float* mw   = (const float*)d_in[13];
  const float* mb   = (const float*)d_in[14];
  const float* w0 = (const float*)d_in[15]; const float* b0 = (const float*)d_in[16];
  const float* w1 = (const float*)d_in[17]; const float* b1 = (const float*)d_in[18];
  const float* w2 = (const float*)d_in[19]; const float* b2 = (const float*)d_in[20];
  const float* w3 = (const float*)d_in[21]; const float* b3 = (const float*)d_in[22];
  const float* w4 = (const float*)d_in[23]; const float* b4 = (const float*)d_in[24];

  // Workspace layout (bf16 activations, K padded to multiples of 32)
  char* ws = (char*)d_ws;
  unsigned short* X  = (unsigned short*)(ws + 0);          // 512 x 20064
  unsigned short* H1 = (unsigned short*)(ws + 20545536);   // 512 x 5024
  unsigned short* H2 = (unsigned short*)(ws + 25690112);   // 512 x 1280
  unsigned short* H3 = (unsigned short*)(ws + 27000832);   // 512 x 320
  unsigned short* H4 = (unsigned short*)(ws + 27328512);   // 512 x 96
  float* out = (float*)d_out;                              // 512 x 2

  // Stage 1: fused front-end
  build_x_kernel<<<dim3(79, 512), 256, 0, stream>>>(
      expr, cc, cnc, bw1, bb1, bw2, bb2, gw, gb,
      race, eth, inter, prot, mw, mb, X);

  // Stage 2..5: WMMA GEMM chain (grid.x = M panels so weights dedupe in L2)
  gemm_bf16_relu<20038, 20064, 5009, 5024><<<dim3(4, 40), 256, 0, stream>>>(X,  w0, b0, H1);
  gemm_bf16_relu<5009, 5024, 1252, 1280><<<dim3(4, 10), 256, 0, stream>>>(H1, w1, b1, H2);
  gemm_bf16_relu<1252, 1280, 313, 320><<<dim3(4, 3),  256, 0, stream>>>(H2, w2, b2, H3);
  gemm_bf16_relu<313, 320, 78, 96><<<dim3(4, 1),  256, 0, stream>>>(H3, w3, b3, H4);

  // Stage 6: tiny output layer, fp32
  final_layer_kernel<<<dim3(4), 256, 0, stream>>>(H4, w4, b4, out);
}
